// DLRM_29772713296020
// MI455X (gfx1250) — matrix operand order repaired and checked by reading, not compiled, not run on hardware
//
#include <hip/hip_runtime.h>
#include <hip/hip_bf16.h>
#include <math.h>

typedef unsigned short ushort_t;
typedef __attribute__((ext_vector_type(16))) __bf16        v16bf;
typedef __attribute__((ext_vector_type(16))) unsigned short v16u;
typedef __attribute__((ext_vector_type(8)))  float          v8f;

#define BATCH      16384
#define NUM_TABLES 26
#define VOCAB      100000
#define M_SPA      64
#define H_COLS     416   /* 64 + 351, padded to 416 */

union BFCast { v16u u; v16bf b; };

__device__ __forceinline__ unsigned f2u(float f) { union { float f; unsigned u; } c; c.f = f; return c.u; }
__device__ __forceinline__ float   u2f(unsigned u) { union { unsigned u; float f; } c; c.u = u; return c.f; }
__device__ __forceinline__ ushort_t f2bf(float f) {
    unsigned x = f2u(f);
    unsigned r = x + 0x7FFFu + ((x >> 16) & 1u);   // round-to-nearest-even
    return (ushort_t)(r >> 16);
}
__device__ __forceinline__ float bf2f(ushort_t b) { return u2f(((unsigned)b) << 16); }

// ---------------------------------------------------------------------------
// Tiled GEMM: C[M,N](bf16) = act(A[M,K] @ B[K,N] + bias), B is f32 row-major.
// BM=128, BN=64, BK=32. 256 threads = 8 wave32s; wave w owns rows w*16..w*16+15
// across all 64 tile columns (4 x v_wmma_f32_16x16x32_bf16 per K-step).
// ---------------------------------------------------------------------------
#define AS_STR 40   /* 80B row stride: 20-bank stagger, conflict-free 16B reads */
#define BS_STR 40

template<bool A_IS_F32>
__global__ __launch_bounds__(256)
void gemm_bias_act(const void* __restrict__ Araw, int lda, int K,
                   const float* __restrict__ B, int N,
                   const float* __restrict__ bias,
                   ushort_t* __restrict__ C, int relu)
{
    __shared__ ushort_t As[128 * AS_STR];
    __shared__ ushort_t Bt[64  * BS_STR];   // stored transposed: Bt[n][k]

    const int tid  = threadIdx.x;
    const int lane = tid & 31;
    const int wave = tid >> 5;
    const int hl   = (lane < 16) ? 0 : 1;   // lane half
    const int l16  = lane & 15;
    const int m0   = blockIdx.y * 128;
    const int n0   = blockIdx.x * 64;

    v8f acc[4];
    #pragma unroll
    for (int j = 0; j < 4; ++j) acc[j] = (v8f){0.f,0.f,0.f,0.f,0.f,0.f,0.f,0.f};

    const int nk = (K + 31) >> 5;
    for (int kt = 0; kt < nk; ++kt) {
        const int kk = kt << 5;

        // --- stage A tile (128x32) as bf16, zero-pad K ---
        for (int idx = tid; idx < 128 * 32; idx += 256) {
            int r = idx >> 5, k = idx & 31;
            int gk = kk + k;
            float v = 0.f;
            if (gk < K) {
                size_t off = (size_t)(m0 + r) * lda + gk;
                v = A_IS_F32 ? ((const float*)Araw)[off]
                             : bf2f(((const ushort_t*)Araw)[off]);
            }
            As[r * AS_STR + k] = A_IS_F32 ? f2bf(v) : (ushort_t)(f2u(v) >> 16);
        }
        // --- stage B tile (32x64) transposed into Bt[n][k] ---
        for (int idx = tid; idx < 64 * 32; idx += 256) {
            int n = idx >> 5, k = idx & 31;
            int gk = kk + k;
            float v = (gk < K) ? B[(size_t)gk * N + (n0 + n)] : 0.f;
            Bt[n * BS_STR + k] = f2bf(v);
        }
        __syncthreads();

        // --- A fragment: row = wave*16 + l16; contiguous k = hl*16 + e layout ---
        BFCast af;
        {
            const ushort_t* ar = &As[(wave * 16 + l16) * AS_STR];
            const int kb = hl * 16;
            #pragma unroll
            for (int e = 0; e < 8; ++e) {
                af.u[e]     = ar[kb + e];        // K = kb..kb+7
                af.u[8 + e] = ar[kb + 8 + e];    // K = kb+8..kb+15
            }
        }
        // --- 4 B fragments + WMMA ---
        const int kb2 = hl * 16;
        #pragma unroll
        for (int j = 0; j < 4; ++j) {
            BFCast bf;
            const ushort_t* br = &Bt[(j * 16 + l16) * BS_STR];
            #pragma unroll
            for (int e = 0; e < 16; ++e) bf.u[e] = br[kb2 + e];
            acc[j] = __builtin_amdgcn_wmma_f32_16x16x32_bf16(
                         false, af.b, false, bf.b, (short)0, acc[j], false, false);
        }
        __syncthreads();
    }

    // --- epilogue: bias + optional ReLU, bf16 store (C layout: M = hl*8+v) ---
    #pragma unroll
    for (int j = 0; j < 4; ++j) {
        const int n = n0 + j * 16 + l16;
        const float bi = bias[n];
        #pragma unroll
        for (int v = 0; v < 8; ++v) {
            const int m = m0 + wave * 16 + hl * 8 + v;
            float x = acc[j][v] + bi;
            if (relu) x = fmaxf(x, 0.f);
            C[(size_t)m * N + n] = f2bf(x);
        }
    }
}

// ---------------------------------------------------------------------------
// Interaction: per sample gather 26 emb rows + d into T (32x64 bf16, rows
// 27..31 zero), Z = T.T^T lower-tri blocks via 6 WMMAs, emit h = [d | tril(Z)].
// One wave per sample, 4 samples per 128-thread block.
// ---------------------------------------------------------------------------
#define IW 4
#define T_STR 72    /* 144B row stride: 36-bank stagger, conflict-free */

__global__ __launch_bounds__(128)
void interact_kernel(const int* __restrict__ sidx,
                     const float* __restrict__ emb,
                     const ushort_t* __restrict__ dbf,
                     ushort_t* __restrict__ h)
{
    __shared__ ushort_t T[IW][32][T_STR];
    __shared__ float    Z[IW][3][16][17];

    const int lane = threadIdx.x & 31;
    const int wave = threadIdx.x >> 5;
    const int hl   = (lane < 16) ? 0 : 1;
    const int l16  = lane & 15;
    const int b    = blockIdx.x * IW + wave;

    // prefetch gathered rows (global_prefetch_b8)
    if (lane < NUM_TABLES) {
        int ridx = sidx[b * NUM_TABLES + lane];
        __builtin_prefetch(&emb[((size_t)lane * VOCAB + ridx) * M_SPA], 0, 1);
    }

    // stage T: rows 0..25 = gathered emb rows (f32->bf16), row 26 = d (bf16)
    for (int r = 0; r < NUM_TABLES; ++r) {
        int ridx = sidx[b * NUM_TABLES + r];
        const float* src = &emb[((size_t)r * VOCAB + ridx) * M_SPA];
        float x0 = src[2 * lane], x1 = src[2 * lane + 1];
        T[wave][r][2 * lane]     = f2bf(x0);
        T[wave][r][2 * lane + 1] = f2bf(x1);
    }
    {
        const ushort_t* dr = &dbf[(size_t)b * M_SPA];
        T[wave][26][2 * lane]     = dr[2 * lane];
        T[wave][26][2 * lane + 1] = dr[2 * lane + 1];
    }
    #pragma unroll
    for (int r = 27; r < 32; ++r) {
        T[wave][r][2 * lane] = 0; T[wave][r][2 * lane + 1] = 0;
    }

    // Z blocks (0,0),(1,0),(1,1): K=64 in two 16x16x32 steps.
    v8f a00 = (v8f){0,0,0,0,0,0,0,0}, a10 = a00, a11 = a00;
    const int kb  = hl * 16;
    const int kb2 = hl * 16;
    #pragma unroll
    for (int kk = 0; kk < 64; kk += 32) {
        BFCast fa0, fa1, fb0, fb1;
        const ushort_t* r0 = T[wave][l16];        // rows 0..15
        const ushort_t* r1 = T[wave][16 + l16];   // rows 16..31
        #pragma unroll
        for (int e = 0; e < 8; ++e) {
            fa0.u[e]     = r0[kk + kb + e];
            fa0.u[8 + e] = r0[kk + kb + 8 + e];
            fa1.u[e]     = r1[kk + kb + e];
            fa1.u[8 + e] = r1[kk + kb + 8 + e];
        }
        #pragma unroll
        for (int e = 0; e < 16; ++e) {            // B = T^T => Bt rows are T rows
            fb0.u[e] = r0[kk + kb2 + e];
            fb1.u[e] = r1[kk + kb2 + e];
        }
        a00 = __builtin_amdgcn_wmma_f32_16x16x32_bf16(false, fa0.b, false, fb0.b, (short)0, a00, false, false);
        a10 = __builtin_amdgcn_wmma_f32_16x16x32_bf16(false, fa1.b, false, fb0.b, (short)0, a10, false, false);
        a11 = __builtin_amdgcn_wmma_f32_16x16x32_bf16(false, fa1.b, false, fb1.b, (short)0, a11, false, false);
    }

    // spill Z blocks to LDS for tril extraction
    #pragma unroll
    for (int v = 0; v < 8; ++v) {
        const int m = hl * 8 + v;
        Z[wave][0][m][l16] = a00[v];
        Z[wave][1][m][l16] = a10[v];
        Z[wave][2][m][l16] = a11[v];
    }

    // h = [ d (64) | tril(Z,-1) (351) | pad (1) ]
    ushort_t* hr = &h[(size_t)b * H_COLS];
    {
        const ushort_t* dr = &dbf[(size_t)b * M_SPA];
        hr[2 * lane]     = dr[2 * lane];
        hr[2 * lane + 1] = dr[2 * lane + 1];
    }
    for (int t = lane; t < 351; t += 32) {
        int i = (int)((1.0f + sqrtf(1.0f + 8.0f * (float)t)) * 0.5f);
        while (i * (i - 1) / 2 > t) --i;
        while ((i + 1) * i / 2 <= t) ++i;
        const int j = t - i * (i - 1) / 2;
        float val;
        if (i < 16)       val = Z[wave][0][i][j];
        else if (j < 16)  val = Z[wave][1][i - 16][j];
        else              val = Z[wave][2][i - 16][j - 16];
        hr[64 + t] = f2bf(val);
    }
    if (lane == 0) hr[415] = 0;
}

// ---------------------------------------------------------------------------
// Final 256 -> 1 layer + sigmoid. One wave per sample, shfl_xor reduction.
// ---------------------------------------------------------------------------
__global__ __launch_bounds__(256)
void final_layer(const ushort_t* __restrict__ h2,
                 const float* __restrict__ Wt2, const float* __restrict__ bt2,
                 float* __restrict__ out)
{
    const int lane = threadIdx.x & 31;
    const int wave = threadIdx.x >> 5;
    const int b    = blockIdx.x * 8 + wave;
    const ushort_t* row = &h2[(size_t)b * 256];
    float s = 0.f;
    #pragma unroll
    for (int c = lane; c < 256; c += 32) s += bf2f(row[c]) * Wt2[c];
    #pragma unroll
    for (int off = 16; off >= 1; off >>= 1) s += __shfl_xor(s, off, 32);
    if (lane == 0) out[b] = 1.f / (1.f + expf(-(s + bt2[0])));
}

// ---------------------------------------------------------------------------
extern "C" void kernel_launch(void* const* d_in, const int* in_sizes, int n_in,
                              void* d_out, int out_size, void* d_ws, size_t ws_size,
                              hipStream_t stream)
{
    const float* dense = (const float*)d_in[0];
    const int*   sidx  = (const int*)  d_in[1];
    const float* emb   = (const float*)d_in[2];
    const float* Wb0   = (const float*)d_in[3];
    const float* bb0   = (const float*)d_in[4];
    const float* Wb1   = (const float*)d_in[5];
    const float* bb1   = (const float*)d_in[6];
    const float* Wb2   = (const float*)d_in[7];
    const float* bb2   = (const float*)d_in[8];
    const float* Wt0   = (const float*)d_in[9];
    const float* bt0   = (const float*)d_in[10];
    const float* Wt1   = (const float*)d_in[11];
    const float* bt1   = (const float*)d_in[12];
    const float* Wt2   = (const float*)d_in[13];
    const float* bt2   = (const float*)d_in[14];
    float* out = (float*)d_out;

    // workspace layout (bf16 activations); h1/h2 alias act0/act1 (disjoint lifetimes)
    char* ws = (char*)d_ws;
    ushort_t* act0 = (ushort_t*)(ws);                                   // 16384x512
    ushort_t* act1 = (ushort_t*)(ws + (size_t)BATCH * 512 * 2);         // 16384x256
    ushort_t* dbf  = (ushort_t*)(ws + (size_t)BATCH * (512 + 256) * 2); // 16384x64
    ushort_t* hbuf = (ushort_t*)(ws + (size_t)BATCH * (512 + 256 + 64) * 2); // 16384x416
    ushort_t* h1   = act0;
    ushort_t* h2   = act1;

    const dim3 blk(256);
    // bottom MLP: 13 -> 512 -> 256 -> 64 (ReLU)
    gemm_bias_act<true ><<<dim3(512 / 64, 128), blk, 0, stream>>>(dense, 13,  13,  Wb0, 512, bb0, act0, 1);
    gemm_bias_act<false><<<dim3(256 / 64, 128), blk, 0, stream>>>(act0, 512, 512,  Wb1, 256, bb1, act1, 1);
    gemm_bias_act<false><<<dim3( 64 / 64, 128), blk, 0, stream>>>(act1, 256, 256,  Wb2,  64, bb2, dbf,  1);
    // gather + pairwise interaction -> h (16384 x 416)
    interact_kernel<<<dim3(BATCH / IW), dim3(128), 0, stream>>>(sidx, emb, dbf, hbuf);
    // top MLP: 415 -> 512 -> 256 (ReLU), then 256 -> 1 + sigmoid
    gemm_bias_act<false><<<dim3(512 / 64, 128), blk, 0, stream>>>(hbuf, H_COLS, 415, Wt0, 512, bt0, h1, 1);
    gemm_bias_act<false><<<dim3(256 / 64, 128), blk, 0, stream>>>(h1,   512,    512, Wt1, 256, bt1, h2, 1);
    final_layer<<<dim3(BATCH / 8), blk, 0, stream>>>(h2, Wt2, bt2, out);
}
